// AutoCorrelation_24240795419432
// MI455X (gfx1250) — compile-verified, matrix operand order checked by reading, not run on hardware
//
#include <hip/hip_runtime.h>

// ---------------------------------------------------------------------------
// Problem constants (from reference): B=8, L=2048, D=512, H=8, DEPTH=64
// ---------------------------------------------------------------------------
#define BATCH 8
#define SEQ   2048
#define DMODEL 512
#define TOPK  15

typedef __attribute__((ext_vector_type(16))) __bf16 v16bf;
typedef __attribute__((ext_vector_type(8)))  float  v8f;

// b128 element type exactly as the builtin expects it (GCC vector of 4 ints).
typedef int b128_t __attribute__((vector_size(16)));
typedef __attribute__((address_space(1))) b128_t* gas_b128;   // global
typedef __attribute__((address_space(3))) b128_t* las_b128;   // LDS

#ifndef __has_builtin
#define __has_builtin(x) 0
#endif
#if __has_builtin(__builtin_amdgcn_global_load_async_to_lds_b128)
#define HAVE_ASYNC_LDS 1
#else
#define HAVE_ASYNC_LDS 0
#endif

// Copy 16 bytes global -> LDS. Async path uses the ASYNCcnt-tracked
// GLOBAL_LOAD_ASYNC_TO_LDS_B128 (CDNA5); fallback is a plain staged copy.
__device__ __forceinline__ void copy16_g2l(const float* __restrict__ g, float* l)
{
#if HAVE_ASYNC_LDS
    __builtin_amdgcn_global_load_async_to_lds_b128(
        (gas_b128)g,            // addrspacecast generic->global, drops const
        (las_b128)l,            // addrspacecast generic->LDS
        /*imm offset=*/0, /*cpol=*/0);
#else
    *(float4*)l = *(const float4*)g;
#endif
}

__device__ __forceinline__ void async_wait_all()
{
#if HAVE_ASYNC_LDS
#if __has_builtin(__builtin_amdgcn_s_wait_asynccnt)
    __builtin_amdgcn_s_wait_asynccnt(0);
#else
    asm volatile("s_wait_asynccnt 0x0" ::: "memory");
#endif
#endif
}

// ---------------------------------------------------------------------------
// WMMA GEMM:  Y = X @ W + bias
//   X: [M=16384, K=512] fp32 row-major (M = b*SEQ + t)
//   W: [K=512, N=512]  fp32 row-major
//   TRANSPOSED=true : store Y[(b*512+n)*SEQ + t]   ([B,H,depth,L] layout)
//   TRANSPOSED=false: store Y[m*512 + n]           (row-major [B,L,D])
// One wave computes a 16x16 tile via 16 x v_wmma_f32_16x16x32_bf16.
// The W tile is staged into LDS pre-swizzled into *fragment order* so each
// B fragment is one aligned 32-byte LDS read per k-step.
// grid = (N/16 = 32, Mtiles/8 = 128), block = 256 (8 waves share the W tile).
// ---------------------------------------------------------------------------
template <bool TRANSPOSED>
__global__ __launch_bounds__(256)
void gemm16_wmma(const float* __restrict__ X, const float* __restrict__ W,
                 const float* __restrict__ bias, float* __restrict__ Y)
{
    // Fragment-order W tile: lds_frag[kb*512 + lane*16 + j], kb = k0/32.
    __shared__ __align__(32) __bf16 lds_frag[16 * 512];   // 16 KB

    const int n0     = blockIdx.x * 16;
    const int wave   = threadIdx.x >> 5;
    const int lane   = threadIdx.x & 31;
    const int m0     = (blockIdx.y * 8 + wave) * 16;

    // Stage W in fragment order:
    //   element (kb, l, j): K = kb*32 + ((j<8) ? (l>>4)*8 + j : 8 + (l>>4)*8 + j)
    //                       N = n0 + (l & 15)
    for (int e = threadIdx.x; e < 16 * 512; e += 256) {
        const int kb = e >> 9;
        const int l  = (e >> 4) & 31;
        const int j  = e & 15;
        const int hf = l >> 4;
        const int ko = (j < 8) ? (hf * 8 + j) : (8 + hf * 8 + j);
        lds_frag[e] = (__bf16)W[(kb * 32 + ko) * DMODEL + n0 + (l & 15)];
    }
    __syncthreads();

    const int rowsel = lane & 15;   // M row (A) / N col (B,C,D) held by this lane
    const int half   = lane >> 4;   // K-interleave selector

    v8f c = {};  // fp32 accumulator

    for (int k0 = 0; k0 < DMODEL; k0 += 32) {
        // ---- A fragment: 16x32 bf16, lane holds row M = rowsel.
        const float* ap = X + (size_t)(m0 + rowsel) * DMODEL + k0 + half * 8;
        __builtin_prefetch(ap + 32, 0, 3);          // global_prefetch for next k-block
        v16bf a;
        #pragma unroll
        for (int j = 0; j < 8; ++j) {
            a[j]     = (__bf16)ap[j];
            a[8 + j] = (__bf16)ap[16 + j];
        }
        // ---- B fragment: one aligned 32-byte LDS read in fragment order.
        const v16bf b = ((const v16bf*)lds_frag)[(k0 >> 5) * 32 + lane];

        c = __builtin_amdgcn_wmma_f32_16x16x32_bf16(
                false, a, false, b, (short)0, c, false, false);
    }

    // Bias depends only on N.
    const float bv = bias[n0 + rowsel];
    #pragma unroll
    for (int r = 0; r < 8; ++r) c[r] += bv;

    // ---- Store. C/D layout: lane -> N = rowsel; VGPR r -> M = r + 8*half.
    if (TRANSPOSED) {
        const int b_idx = m0 >> 11;
        const int t0    = m0 & (SEQ - 1);
        const int n     = n0 + rowsel;
        float* dst = Y + ((size_t)(b_idx * DMODEL + n)) * SEQ + t0 + half * 8;
        ((float4*)dst)[0] = make_float4(c[0], c[1], c[2], c[3]);
        ((float4*)dst)[1] = make_float4(c[4], c[5], c[6], c[7]);
    } else {
        const int n = n0 + rowsel;
        float* dst = Y + (size_t)(m0 + half * 8) * DMODEL + n;
        #pragma unroll
        for (int r = 0; r < 8; ++r) dst[(size_t)r * DMODEL] = c[r];
    }
}

// ---------------------------------------------------------------------------
// Fused circular correlation + top-15 + softmax + delay aggregation.
// One block (256 threads) per series (4096 series), series length 2048.
//   R[tau] = sum_t q[(t+tau)%L] * k[t]
// Thread owns tau = tid*8 .. tid*8+7 and keeps a 16-wide sliding register
// window of q, so per 64 FMAs it does one 32B q LDS read + one broadcast
// 32B k read. sq2 is the series pre-wrapped to length 4104 so no modulo is
// needed anywhere (max index used: 4095).
// ---------------------------------------------------------------------------
__global__ __launch_bounds__(256)
void autocorr_kernel(const float* __restrict__ qt, const float* __restrict__ kt,
                     float* __restrict__ out_pre)
{
    __shared__ __align__(16) float sq2[4104];   // q series, wrapped (16.4 KB)
    __shared__ __align__(16) float sk[SEQ];     // k series (8 KB)
    __shared__ float sR[SEQ];
    __shared__ float red_v[256];
    __shared__ int   red_i[256];
    __shared__ float topv[TOPK];
    __shared__ int   topi[TOPK];

    const int s   = blockIdx.x;                  // series = b*512 + n
    const int tid = threadIdx.x;
    const float* qs = qt + (size_t)s * SEQ;
    const float* ks = kt + (size_t)s * SEQ;

    // Stage via async global->LDS copies (16B chunks), then wait + barrier.
    for (int x = tid * 4; x < 4104; x += 1024)
        copy16_g2l(qs + (x & (SEQ - 1)), &sq2[x]);
    for (int x = tid * 4; x < SEQ; x += 1024)
        copy16_g2l(ks + x, &sk[x]);
    async_wait_all();
    __syncthreads();

    // ---- correlation with sliding window
    const int tau0 = tid * 8;
    float w[16];
    #pragma unroll
    for (int j = 0; j < 16; ++j) w[j] = sq2[tau0 + j];

    float acc[8] = {0.f, 0.f, 0.f, 0.f, 0.f, 0.f, 0.f, 0.f};
    for (int t = 0; t < SEQ; t += 8) {
        float kv[8];
        #pragma unroll
        for (int d = 0; d < 8; ++d) kv[d] = sk[t + d];
        #pragma unroll
        for (int d = 0; d < 8; ++d) {
            #pragma unroll
            for (int i = 0; i < 8; ++i)
                acc[i] = fmaf(w[d + i], kv[d], acc[i]);
        }
        #pragma unroll
        for (int j = 0; j < 8; ++j) w[j] = w[j + 8];
        if (t + 8 < SEQ) {
            #pragma unroll
            for (int j = 0; j < 8; ++j) w[8 + j] = sq2[t + 16 + tau0 + j];
        }
    }
    #pragma unroll
    for (int i = 0; i < 8; ++i) sR[tau0 + i] = acc[i];
    __syncthreads();

    // ---- top-15 via repeated argmax reduction
    for (int sel = 0; sel < TOPK; ++sel) {
        float best = -3.0e38f; int bi = 0;
        #pragma unroll
        for (int i = 0; i < 8; ++i) {
            const float v = sR[tau0 + i];
            if (v > best) { best = v; bi = tau0 + i; }
        }
        red_v[tid] = best; red_i[tid] = bi;
        __syncthreads();
        for (int off = 128; off > 0; off >>= 1) {
            if (tid < off) {
                if (red_v[tid + off] > red_v[tid]) {
                    red_v[tid] = red_v[tid + off];
                    red_i[tid] = red_i[tid + off];
                }
            }
            __syncthreads();
        }
        if (tid == 0) {
            topv[sel] = red_v[0];
            topi[sel] = red_i[0];
            sR[red_i[0]] = -3.0e38f;
        }
        __syncthreads();
    }

    // ---- softmax over the 15 weights (redundant per thread)
    float wgt[TOPK];
    float m = -3.0e38f;
    #pragma unroll
    for (int i = 0; i < TOPK; ++i) m = fmaxf(m, topv[i]);
    float ssum = 0.f;
    #pragma unroll
    for (int i = 0; i < TOPK; ++i) { wgt[i] = __expf(topv[i] - m); ssum += wgt[i]; }
    const float inv = 1.f / ssum;
    #pragma unroll
    for (int i = 0; i < TOPK; ++i) wgt[i] *= inv;

    // ---- delay aggregation of q (sq2 is pre-wrapped: max index 4094)
    const int b_idx = s >> 9;
    const int n     = s & 511;
    #pragma unroll
    for (int j = 0; j < 8; ++j) {
        const int t = tid + j * 256;
        float a = 0.f;
        #pragma unroll
        for (int i = 0; i < TOPK; ++i)
            a = fmaf(sq2[t + topi[i]], wgt[i], a);
        out_pre[((size_t)(b_idx << 11) + t) * DMODEL + n] = a;
    }
}

// ---------------------------------------------------------------------------
// Launcher. Inputs (fp32): q,k,v, Wq,bq, Wk,bk, Wv,bv, Wo,bo.
// v/Wv/bv are dead in the reference -> skipped.
// Workspace: qt (32MB) | kt (32MB) | out_pre (32MB) = 96MB.
// ---------------------------------------------------------------------------
extern "C" void kernel_launch(void* const* d_in, const int* in_sizes, int n_in,
                              void* d_out, int out_size, void* d_ws, size_t ws_size,
                              hipStream_t stream)
{
    const float* q  = (const float*)d_in[0];
    const float* k  = (const float*)d_in[1];
    const float* Wq = (const float*)d_in[3];
    const float* bq = (const float*)d_in[4];
    const float* Wk = (const float*)d_in[5];
    const float* bk = (const float*)d_in[6];
    const float* Wo = (const float*)d_in[9];
    const float* bo = (const float*)d_in[10];
    float* out = (float*)d_out;

    const size_t planesz = (size_t)BATCH * DMODEL * SEQ;   // 8M floats
    float* qt = (float*)d_ws;
    float* kt = qt + planesz;
    float* op = kt + planesz;

    dim3 grid_g(DMODEL / 16, (BATCH * SEQ / 16) / 8);      // (32, 128)
    dim3 blk(256);

    gemm16_wmma<true ><<<grid_g, blk, 0, stream>>>(q,  Wq, bq, qt);
    gemm16_wmma<true ><<<grid_g, blk, 0, stream>>>(k,  Wk, bk, kt);
    autocorr_kernel   <<<BATCH * DMODEL, blk, 0, stream>>>(qt, kt, op);
    gemm16_wmma<false><<<grid_g, blk, 0, stream>>>(op, Wo, bo, out);
}
